// GCN_58222576664682
// MI455X (gfx1250) — matrix operand order repaired and checked by reading, not compiled, not run on hardware
//
#include <hip/hip_runtime.h>
#include <hip/hip_bf16.h>
#include <math.h>

typedef __attribute__((ext_vector_type(16))) __bf16 v16bf;
typedef __attribute__((ext_vector_type(8)))  float  v8f;

#define HDIM 128
#define GDIM 512
#define BN_EPS 1e-5f

__device__ __forceinline__ void atomAddF(float* p, float v) {
  (void)__hip_atomic_fetch_add(p, v, __ATOMIC_RELAXED, __HIP_MEMORY_SCOPE_AGENT);
}

// ---------------- generic fill ----------------
__global__ void k_fill(float* __restrict__ p, float v, int n) {
  int i = blockIdx.x * blockDim.x + threadIdx.x;
  if (i < n) p[i] = v;
}

// ---------------- AtomEncoder: h[n][c] = sum_j table[x[n][j]+off[j]][c] ----------------
__global__ void k_encode(const int* __restrict__ x, const float* __restrict__ tbl,
                         float* __restrict__ h, int n) {
  int idx = blockIdx.x * blockDim.x + threadIdx.x;
  if (idx >= n * HDIM) return;
  int node = idx >> 7, c = idx & 127;
  const int offs[9] = {0, 119, 123, 135, 147, 157, 163, 169, 171};
  const int* xi = x + node * 9;
  float s = 0.f;
#pragma unroll
  for (int j = 0; j < 9; ++j) s += tbl[(xi[j] + offs[j]) * HDIM + c];
  h[idx] = s;
}

// ---------------- degree (init to 1.0 for self loop, add per edge on col) ----------------
__global__ void k_deg(const int* __restrict__ col, float* __restrict__ deg, int E) {
  int e = blockIdx.x * blockDim.x + threadIdx.x;
  if (e < E) atomAddF(&deg[col[e]], 1.0f);
}

__global__ void k_dis(const float* __restrict__ deg, float* __restrict__ dis, int n) {
  int i = blockIdx.x * blockDim.x + threadIdx.x;
  if (i < n) dis[i] = rsqrtf(deg[i]);  // deg >= 1 always (self loop)
}

__global__ void k_norm(const int* __restrict__ row, const int* __restrict__ col,
                       const float* __restrict__ dis, float* __restrict__ nrm, int E) {
  int e = blockIdx.x * blockDim.x + threadIdx.x;
  if (e < E) nrm[e] = dis[row[e]] * dis[col[e]];
}

// ---------------- GEMM: ht = h @ W, bf16 WMMA with f32 accumulate ----------------
// Block = 256 threads = 8 waves. W (128x128 f32) is converted to bf16 and
// staged once per block in LDS, pre-shuffled into exact B-fragment order so a
// lane's 16-element fragment is one contiguous 32-B vector load
// (lds_v16[(ntile*4+kc)*32 + lane]). Each wave then grid-strides over 16-row
// strips of h, amortizing the W staging across several strips. The strip loop
// bound is wave-uniform, so EXEC is all-1s for every WMMA.
__global__ __launch_bounds__(256) void k_gemm(const float* __restrict__ hin,
                                              const float* __restrict__ W,
                                              float* __restrict__ ht, int nstrips) {
  __shared__ __bf16 Wl[HDIM * HDIM];  // 32 KB
  const int tid = threadIdx.x;
  // Shuffle W[k][ncol] into fragment order:
  //   k = kc*32 + hi*16 + j  (kc=K-chunk, hi=lane half, j=0..15)
  //   ncol = ntile*16 + ncl
  //   dst = (((ntile*4+kc)*2 + hi)*16 + ncl)*16 + j
  for (int e = tid; e < HDIM * HDIM; e += 256) {
    int k = e >> 7, ncol = e & 127;
    int kc = k >> 5, hi = (k >> 4) & 1, j = k & 15;
    int ntile = ncol >> 4, ncl = ncol & 15;
    Wl[((((ntile << 2) | kc) * 2 + hi) * 16 + ncl) * 16 + j] = (__bf16)W[e];
  }
  __syncthreads();

  const v16bf* Wv = (const v16bf*)Wl;
  const int wave = tid >> 5, lane = tid & 31;
  const int m = lane & 15, hi = lane >> 4;
  const int wave0 = blockIdx.x * 8 + wave;
  const int nwaves = gridDim.x * 8;

  for (int strip = wave0; strip < nstrips; strip += nwaves) {
    const float* arow = hin + (size_t)(strip * 16 + m) * HDIM;

    // A fragments (contiguous k order, matching B): lane half hi holds
    //   frag[i]   = A[m][kc*32 + hi*16 + i]       (i = 0..7)
    //   frag[8+i] = A[m][kc*32 + hi*16 + 8 + i]
    v16bf a[4];
#pragma unroll
    for (int kc = 0; kc < 4; ++kc) {
      const float* p0 = arow + kc * 32 + hi * 16;
      const float* p1 = p0 + 8;
#pragma unroll
      for (int i = 0; i < 8; ++i) {
        a[kc][i]     = (__bf16)p0[i];
        a[kc][8 + i] = (__bf16)p1[i];
      }
    }

#pragma unroll
    for (int ntile = 0; ntile < 8; ++ntile) {
      v8f c = {};
#pragma unroll
      for (int kc = 0; kc < 4; ++kc) {
        v16bf b = Wv[((ntile << 2) | kc) * 32 + lane];
        c = __builtin_amdgcn_wmma_f32_16x16x32_bf16(
            /*neg_a=*/false, a[kc], /*neg_b=*/false, b,
            /*c_mod=*/(short)0, c, /*reuse_a=*/false, /*reuse_b=*/false);
      }
      // C layout: VGPR r -> row hi*8 + r, col = ntile*16 + (lane&15)
      float* orow = ht + (size_t)(strip * 16 + hi * 8) * HDIM + ntile * 16 + (lane & 15);
#pragma unroll
      for (int r = 0; r < 8; ++r) orow[(size_t)r * HDIM] = c[r];
    }
  }
}

// Scalar tail for leftover rows (never fires for N % 16 == 0)
__global__ void k_gemm_tail(const float* __restrict__ hin, const float* __restrict__ W,
                            float* __restrict__ ht, int start, int n) {
  int rows = n - start;
  int idx = blockIdx.x * blockDim.x + threadIdx.x;
  if (idx >= rows * HDIM) return;
  int r = start + (idx >> 7), c = idx & 127;
  float s = 0.f;
  for (int k = 0; k < HDIM; ++k) s += hin[(size_t)r * HDIM + k] * W[k * HDIM + c];
  ht[(size_t)r * HDIM + c] = s;
}

// ---------------- agg = ht * selfnorm + bias ----------------
__global__ void k_agg_init(const float* __restrict__ ht, const float* __restrict__ dis,
                           const float* __restrict__ bias, float* __restrict__ agg, int n) {
  int idx = blockIdx.x * blockDim.x + threadIdx.x;
  if (idx >= n * HDIM) return;
  int node = idx >> 7, c = idx & 127;
  float d = dis[node];
  agg[idx] = ht[idx] * d * d + bias[c];
}

// ---------------- agg[col[e]] += ht[row[e]] * norm[e] ----------------
__global__ void k_scatter(const int* __restrict__ row, const int* __restrict__ col,
                          const float* __restrict__ nrm, const float* __restrict__ ht,
                          float* __restrict__ agg, int E) {
  int idx = blockIdx.x * blockDim.x + threadIdx.x;
  if (idx >= E * HDIM) return;
  int e = idx >> 7, c = idx & 127;
  atomAddF(&agg[(size_t)col[e] * HDIM + c], ht[(size_t)row[e] * HDIM + c] * nrm[e]);
}

// ---------------- BN stats: per-channel sum & sumsq ----------------
__global__ void k_stats(const float* __restrict__ agg, float* __restrict__ stats, int n) {
  int gid = blockIdx.x * blockDim.x + threadIdx.x;  // blockDim = 128
  int c = gid & 127;
  int stride = gridDim.x * blockDim.x;  // multiple of 128 -> c fixed per thread
  float s = 0.f, s2 = 0.f;
  for (int i = gid; i < n * HDIM; i += stride) {
    float v = agg[i];
    s += v;
    s2 += v * v;
  }
  atomAddF(&stats[c], s);
  atomAddF(&stats[HDIM + c], s2);
}

__global__ void k_stats_fin(float* __restrict__ stats, const float* __restrict__ gamma,
                            const float* __restrict__ beta, int n) {
  int c = threadIdx.x;  // 128 threads
  float inv = 1.0f / (float)n;
  float mean = stats[c] * inv;
  float var = stats[HDIM + c] * inv - mean * mean;
  float sc = gamma[c] * rsqrtf(var + BN_EPS);
  stats[c] = sc;
  stats[HDIM + c] = beta[c] - mean * sc;
}

// ---------------- h += act(agg*scale + shift) ----------------
__global__ void k_bn(const float* __restrict__ agg, const float* __restrict__ stats,
                     float* __restrict__ h, int n, int relu) {
  int idx = blockIdx.x * blockDim.x + threadIdx.x;
  if (idx >= n * HDIM) return;
  int c = idx & 127;
  float v = agg[idx] * stats[c] + stats[HDIM + c];
  if (relu) v = fmaxf(v, 0.f);
  h[idx] += v;
}

// ---------------- mean pooling + linear + sigmoid ----------------
__global__ void k_pool(const int* __restrict__ batch, const float* __restrict__ h,
                       float* __restrict__ pooled, float* __restrict__ counts, int n) {
  int idx = blockIdx.x * blockDim.x + threadIdx.x;
  if (idx >= n * HDIM) return;
  int node = idx >> 7, c = idx & 127;
  int g = batch[node];
  atomAddF(&pooled[(size_t)g * HDIM + c], h[idx]);
  if (c == 0) atomAddF(&counts[g], 1.0f);
}

__global__ void k_readout(const float* __restrict__ pooled, const float* __restrict__ counts,
                          const float* __restrict__ lw, const float* __restrict__ lb,
                          float* __restrict__ out) {
  __shared__ float red[HDIM];
  int g = blockIdx.x, c = threadIdx.x;
  float cnt = fmaxf(counts[g], 1.0f);
  red[c] = (pooled[(size_t)g * HDIM + c] / cnt) * lw[c];
  __syncthreads();
  for (int s = 64; s > 0; s >>= 1) {
    if (c < s) red[c] += red[c + s];
    __syncthreads();
  }
  if (c == 0) out[g] = 1.0f / (1.0f + expf(-(red[0] + lb[0])));
}

// ---------------- host launcher ----------------
extern "C" void kernel_launch(void* const* d_in, const int* in_sizes, int n_in,
                              void* d_out, int out_size, void* d_ws, size_t ws_size,
                              hipStream_t stream) {
  const int*   x     = (const int*)d_in[0];
  const int*   ei    = (const int*)d_in[1];
  const int*   batch = (const int*)d_in[2];
  const float* atbl  = (const float*)d_in[3];
  const float* cW    = (const float*)d_in[4];
  const float* cb    = (const float*)d_in[5];
  const float* gam   = (const float*)d_in[6];
  const float* bet   = (const float*)d_in[7];
  const float* lw    = (const float*)d_in[8];
  const float* lb    = (const float*)d_in[9];
  float*       out   = (float*)d_out;

  const int N = in_sizes[0] / 9;
  const int E = in_sizes[1] / 2;
  const int L = in_sizes[4] / (HDIM * HDIM);
  const int* row = ei;        // edge_index[0]
  const int* col = ei + E;    // edge_index[1]

  float* ws     = (float*)d_ws;
  float* h      = ws;
  float* ht     = h + (size_t)N * HDIM;
  float* agg    = ht + (size_t)N * HDIM;
  float* deg    = agg + (size_t)N * HDIM;
  float* dis    = deg + N;
  float* nrm    = dis + N;
  float* stats  = nrm + E;
  float* pooled = stats + 2 * HDIM;
  float* counts = pooled + (size_t)GDIM * HDIM;

  const int NH = N * HDIM;
  const int EH = E * HDIM;
  auto cdiv = [](int a, int b) { return (a + b - 1) / b; };

  // encode + normalization coefficients
  k_encode<<<cdiv(NH, 256), 256, 0, stream>>>(x, atbl, h, N);
  k_fill<<<cdiv(N, 256), 256, 0, stream>>>(deg, 1.0f, N);
  k_deg<<<cdiv(E, 256), 256, 0, stream>>>(col, deg, E);
  k_dis<<<cdiv(N, 256), 256, 0, stream>>>(deg, dis, N);
  k_norm<<<cdiv(E, 256), 256, 0, stream>>>(row, col, dis, nrm, E);

  const int nstrips = N / 16;
  const int rem = N - nstrips * 16;
  int gemm_grid = cdiv(nstrips, 8);
  if (gemm_grid > 256) gemm_grid = 256;  // grid-stride: ~3 strips/wave, 16 MB W traffic/layer

  for (int l = 0; l < L; ++l) {
    k_gemm<<<gemm_grid, 256, 0, stream>>>(h, cW + (size_t)l * HDIM * HDIM, ht, nstrips);
    if (rem > 0)
      k_gemm_tail<<<cdiv(rem * HDIM, 256), 256, 0, stream>>>(
          h, cW + (size_t)l * HDIM * HDIM, ht, nstrips * 16, N);
    k_agg_init<<<cdiv(NH, 256), 256, 0, stream>>>(ht, dis, cb + l * HDIM, agg, N);
    k_scatter<<<cdiv(EH, 256), 256, 0, stream>>>(row, col, nrm, ht, agg, E);
    k_fill<<<1, 256, 0, stream>>>(stats, 0.f, 2 * HDIM);
    k_stats<<<512, 128, 0, stream>>>(agg, stats, N);
    k_stats_fin<<<1, 128, 0, stream>>>(stats, gam + l * HDIM, bet + l * HDIM, N);
    k_bn<<<cdiv(NH, 256), 256, 0, stream>>>(agg, stats, h, N, (l < L - 1) ? 1 : 0);
  }

  k_fill<<<cdiv(GDIM * HDIM + GDIM, 256), 256, 0, stream>>>(pooled, 0.f, GDIM * HDIM + GDIM);
  k_pool<<<cdiv(NH, 256), 256, 0, stream>>>(batch, h, pooled, counts, N);
  k_readout<<<GDIM, HDIM, 0, stream>>>(pooled, counts, lw, lb, out);
}
